// MambaLMFunctional_53901839564951
// MI455X (gfx1250) — compile-verified, matrix operand order checked by reading, not run on hardware
//
#include <hip/hip_runtime.h>
#include <hip/hip_bf16.h>

// ---------------- model constants ----------------
#define D_MODEL 1024
#define N_LAYERS 2
#define VOCAB 32000
#define D_STATE 16
#define D_CONV 4
#define D_INNER 2048
#define DT_RANK 64
#define BATCH 2
#define SEQLEN 1024
#define NTOK (BATCH * SEQLEN)
#define EPS 1e-5f

typedef __attribute__((ext_vector_type(16))) __bf16 v16bf;
typedef __attribute__((ext_vector_type(8)))  float  v8f;
typedef __attribute__((ext_vector_type(4)))  int    i4v;

union FragU { v16bf v; uint4 q[2]; };

#define AS1 __attribute__((address_space(1)))
#define AS3 __attribute__((address_space(3)))

#if __has_builtin(__builtin_amdgcn_global_load_async_to_lds_b128)
#define HAVE_ASYNC_LDS 1
#else
#define HAVE_ASYNC_LDS 0
#endif

__device__ __forceinline__ void wait_async_lds() {
#if HAVE_ASYNC_LDS
#if __has_builtin(__builtin_amdgcn_s_wait_asynccnt)
    __builtin_amdgcn_s_wait_asynccnt(0);
#else
    asm volatile("s_wait_asynccnt 0" ::: "memory");
#endif
#endif
}

__device__ __forceinline__ float silu_f(float z) {
    return z / (1.0f + __expf(-z));
}

// ---------------- cast fp32 -> bf16 ----------------
__global__ void k_cast_bf16(const float* __restrict__ src, __bf16* __restrict__ dst, long n) {
    long i = (long)blockIdx.x * blockDim.x + threadIdx.x;
    long stride = (long)gridDim.x * blockDim.x;
    for (; i < n; i += stride) dst[i] = (__bf16)src[i];
}

// ---------------- embedding gather ----------------
__global__ void k_embed(const int* __restrict__ tokens, const float* __restrict__ embed,
                        float* __restrict__ x) {
    int row = blockIdx.x;                      // 0..NTOK-1
    int tok = tokens[row];
    const float* src = embed + (size_t)tok * D_MODEL;
    float* dst = x + (size_t)row * D_MODEL;
    for (int i = threadIdx.x; i < D_MODEL; i += blockDim.x) dst[i] = src[i];
}

// ---------------- RMSNorm (one block per row) ----------------
__global__ void k_rmsnorm(const float* __restrict__ x, const float* __restrict__ w,
                          float* __restrict__ out, int D) {
    int row = blockIdx.x;
    const float* xr = x + (size_t)row * D;
    __shared__ float red[256];
    float s = 0.f;
    for (int i = threadIdx.x; i < D; i += 256) { float v = xr[i]; s += v * v; }
    red[threadIdx.x] = s;
    __syncthreads();
    for (int off = 128; off > 0; off >>= 1) {
        if (threadIdx.x < off) red[threadIdx.x] += red[threadIdx.x + off];
        __syncthreads();
    }
    float inv = rsqrtf(red[0] / (float)D + EPS);
    float* o = out + (size_t)row * D;
    for (int i = threadIdx.x; i < D; i += 256) o[i] = xr[i] * inv * w[i];
}

// ---------------- bf16 WMMA GEMM: C[M,N] = A[M,K] * W[N,K]^T (+Res) ----------------
// Block: 256 threads / 8 waves -> 64x128 C tile. Wave: 32x32 (2x2 WMMA tiles).
// Double-buffered LDS staging via GLOBAL_LOAD_ASYNC_TO_LDS (ASYNCcnt) when available.
#define LDS_PAD 40   // 32 elems + 8 pad -> 80B row stride, conflict-free b128 reads

__global__ __launch_bounds__(256) void k_gemm_bf16(
    const __bf16* __restrict__ A, const __bf16* __restrict__ Bw,
    const float* Res, float* C,
    int M, int N, int K, int lda, int ldb, int ldc)
{
    __shared__ __align__(16) __bf16 sA[2][64 * LDS_PAD];
    __shared__ __align__(16) __bf16 sB[2][128 * LDS_PAD];

    const int tid  = threadIdx.x;
    const int wave = tid >> 5;
    const int lane = tid & 31;
    const int hw   = lane >> 4;       // K half-split across lane halves
    const int r    = lane & 15;       // row (A) / col (B) within a 16-tile

    const int wm = wave & 1;          // M sub-tile: 0..1  (32 rows each)
    const int wn = wave >> 1;         // N sub-tile: 0..3  (32 cols each)
    const int m_blk = blockIdx.y * 64;
    const int n_blk = blockIdx.x * 128;

    // staging coordinates: each thread copies one 16B segment of A and two of B
    const int crow = tid >> 2;        // 0..63
    const int cseg = (tid & 3) * 8;   // element offset within the 32-wide K chunk

    v8f acc[2][2] = {};

    const int nk = K >> 5;
    int cur = 0;

    auto stage = [&](int buf, int k0) {
        const __bf16* gA  = A  + (size_t)(m_blk + crow) * lda + k0 + cseg;
        const __bf16* gB0 = Bw + (size_t)(n_blk + crow) * ldb + k0 + cseg;
        const __bf16* gB1 = Bw + (size_t)(n_blk + 64 + crow) * ldb + k0 + cseg;
        __bf16* dA  = &sA[buf][crow * LDS_PAD + cseg];
        __bf16* dB0 = &sB[buf][crow * LDS_PAD + cseg];
        __bf16* dB1 = &sB[buf][(64 + crow) * LDS_PAD + cseg];
#if HAVE_ASYNC_LDS
        __builtin_amdgcn_global_load_async_to_lds_b128((AS1 i4v*)gA, (AS3 i4v*)dA, 0, 0);
        if (n_blk + crow < N)
            __builtin_amdgcn_global_load_async_to_lds_b128((AS1 i4v*)gB0, (AS3 i4v*)dB0, 0, 0);
        if (n_blk + 64 + crow < N)
            __builtin_amdgcn_global_load_async_to_lds_b128((AS1 i4v*)gB1, (AS3 i4v*)dB1, 0, 0);
#else
        *(uint4*)dA = *(const uint4*)gA;
        if (n_blk + crow < N)       *(uint4*)dB0 = *(const uint4*)gB0;
        if (n_blk + 64 + crow < N)  *(uint4*)dB1 = *(const uint4*)gB1;
#endif
    };

    stage(0, 0);
    wait_async_lds();
    __syncthreads();

    for (int kc = 0; kc < nk; ++kc) {
        if (kc + 1 < nk) stage(cur ^ 1, (kc + 1) * 32);

        FragU fa0, fa1, fb0, fb1;
        const __bf16* ar = &sA[cur][(wm * 32 + r) * LDS_PAD];
        fa0.q[0] = *(const uint4*)(ar + hw * 8);
        fa0.q[1] = *(const uint4*)(ar + 16 + hw * 8);
        fa1.q[0] = *(const uint4*)(ar + 16 * LDS_PAD + hw * 8);
        fa1.q[1] = *(const uint4*)(ar + 16 * LDS_PAD + 16 + hw * 8);
        const __bf16* br = &sB[cur][(wn * 32 + r) * LDS_PAD];
        fb0.q[0] = *(const uint4*)(br + hw * 8);
        fb0.q[1] = *(const uint4*)(br + 16 + hw * 8);
        fb1.q[0] = *(const uint4*)(br + 16 * LDS_PAD + hw * 8);
        fb1.q[1] = *(const uint4*)(br + 16 * LDS_PAD + 16 + hw * 8);

        acc[0][0] = __builtin_amdgcn_wmma_f32_16x16x32_bf16(false, fa0.v, false, fb0.v,
                                                            (short)0, acc[0][0], false, false);
        acc[0][1] = __builtin_amdgcn_wmma_f32_16x16x32_bf16(false, fa0.v, false, fb1.v,
                                                            (short)0, acc[0][1], false, false);
        acc[1][0] = __builtin_amdgcn_wmma_f32_16x16x32_bf16(false, fa1.v, false, fb0.v,
                                                            (short)0, acc[1][0], false, false);
        acc[1][1] = __builtin_amdgcn_wmma_f32_16x16x32_bf16(false, fa1.v, false, fb1.v,
                                                            (short)0, acc[1][1], false, false);

        wait_async_lds();
        __syncthreads();
        cur ^= 1;
    }

    // C/D layout: lane half selects M base (0 or 8), VGPR i = row offset, r = col
#pragma unroll
    for (int ti = 0; ti < 2; ++ti) {
#pragma unroll
        for (int tj = 0; tj < 2; ++tj) {
            int crow0 = m_blk + wm * 32 + ti * 16 + hw * 8;
            int ccol  = n_blk + wn * 32 + tj * 16 + r;
            if (ccol < N) {
#pragma unroll
                for (int i = 0; i < 8; ++i) {
                    size_t off = (size_t)(crow0 + i) * ldc + ccol;
                    float v = acc[ti][tj][i];
                    if (Res) v += Res[off];
                    C[off] = v;
                }
            }
        }
    }
}

// ---------------- depthwise causal conv (D_CONV=4) + bias + silu ----------------
__global__ void k_conv_silu(const float* __restrict__ xz, const float* __restrict__ cw,
                            const float* __restrict__ cb, float* __restrict__ xs) {
    long i = (long)blockIdx.x * blockDim.x + threadIdx.x;
    if (i >= (long)NTOK * D_INNER) return;
    int d = (int)(i % D_INNER);
    long row = i / D_INNER;                 // b*SEQLEN + t
    int t = (int)(row % SEQLEN);
    long base = row * (2 * D_INNER) + d;
    float acc = cb[d];
#pragma unroll
    for (int j = 0; j < D_CONV; ++j) {
        int tt = t - (D_CONV - 1) + j;
        if (tt >= 0)
            acc += cw[d * D_CONV + j] * xz[base + (long)(tt - t) * (2 * D_INNER)];
    }
    xs[i] = silu_f(acc);
}

// conv cache: (B, D_INNER, D_CONV-1) from last 3 timesteps of xi
__global__ void k_conv_cache(const float* __restrict__ xz, float* __restrict__ cache) {
    int i = blockIdx.x * blockDim.x + threadIdx.x;
    if (i >= BATCH * D_INNER * (D_CONV - 1)) return;
    int j = i % (D_CONV - 1);
    int d = (i / (D_CONV - 1)) % D_INNER;
    int b = i / ((D_CONV - 1) * D_INNER);
    long row = (long)b * SEQLEN + (SEQLEN - (D_CONV - 1) + j);
    cache[i] = xz[row * (2 * D_INNER) + d];
}

// ---------------- delta = softplus(draw + dt_bias) (in place) ----------------
__global__ void k_softplus_bias(float* __restrict__ delta, const float* __restrict__ bias) {
    long i = (long)blockIdx.x * blockDim.x + threadIdx.x;
    if (i >= (long)NTOK * D_INNER) return;
    int d = (int)(i % D_INNER);
    float v = delta[i] + bias[d];
    delta[i] = (v > 20.f) ? v : __logf(1.f + __expf(v));
}

// ---------------- fused selective scan ----------------
__global__ __launch_bounds__(128) void k_scan(
    const float* __restrict__ delta,   // (NTOK, D_INNER)
    const float* __restrict__ xs,      // (NTOK, D_INNER)
    const float* __restrict__ dbl,     // (NTOK, 96) : B at 64, C at 80
    const float* __restrict__ xz,      // (NTOK, 2*D_INNER) : z = second half
    const float* __restrict__ A_log,   // (D_INNER, D_STATE)
    const float* __restrict__ Dp,      // (D_INNER)
    float* __restrict__ y,             // (NTOK, D_INNER)
    float* __restrict__ hs_out)        // (BATCH, D_INNER, D_STATE)
{
    int ch = blockIdx.x * blockDim.x + threadIdx.x;   // 0 .. BATCH*D_INNER-1
    int b = ch / D_INNER;
    int d = ch % D_INNER;

    float Arow[D_STATE];
    float h[D_STATE];
#pragma unroll
    for (int n = 0; n < D_STATE; ++n) {
        Arow[n] = -__expf(A_log[(size_t)d * D_STATE + n]);
        h[n] = 0.f;
    }
    float Dd = Dp[d];

    for (int t = 0; t < SEQLEN; ++t) {
        long row = (long)b * SEQLEN + t;
        float dt   = delta[row * D_INNER + d];
        float xval = xs[row * D_INNER + d];
        const float* bc = dbl + row * (DT_RANK + 2 * D_STATE);
        float acc = 0.f;
#pragma unroll
        for (int n = 0; n < D_STATE; ++n) {
            float Bn = bc[DT_RANK + n];
            float Cn = bc[DT_RANK + D_STATE + n];
            h[n] = __expf(dt * Arow[n]) * h[n] + dt * Bn * xval;
            acc += h[n] * Cn;
        }
        float z = xz[row * (2 * D_INNER) + D_INNER + d];
        y[row * D_INNER + d] = (acc + Dd * xval) * silu_f(z);
    }
#pragma unroll
    for (int n = 0; n < D_STATE; ++n)
        hs_out[((size_t)b * D_INNER + d) * D_STATE + n] = h[n];
}

// ---------------- host side ----------------
static inline dim3 gemm_grid(int M, int N) {
    return dim3((N + 127) / 128, (M + 63) / 64);
}

extern "C" void kernel_launch(void* const* d_in, const int* in_sizes, int n_in,
                              void* d_out, int out_size, void* d_ws, size_t ws_size,
                              hipStream_t stream) {
    (void)in_sizes; (void)n_in; (void)out_size; (void)ws_size;

    const int*   tokens    = (const int*)  d_in[0];
    const float* embed     = (const float*)d_in[1];
    const float* in_proj_w = (const float*)d_in[2];   // (L, 4096, 1024)
    const float* conv_w    = (const float*)d_in[3];   // (L, 2048, 4)
    const float* conv_b    = (const float*)d_in[4];   // (L, 2048)
    const float* x_proj_w  = (const float*)d_in[5];   // (L, 96, 2048)
    const float* dt_proj_w = (const float*)d_in[6];   // (L, 2048, 64)
    const float* dt_proj_b = (const float*)d_in[7];   // (L, 2048)
    const float* A_log     = (const float*)d_in[8];   // (L, 2048, 16)
    const float* D_param   = (const float*)d_in[9];   // (L, 2048)
    const float* out_proj_w= (const float*)d_in[10];  // (L, 1024, 2048)
    const float* rms_w     = (const float*)d_in[11];  // (L, 1024)
    const float* norm_f_w  = (const float*)d_in[12];  // (1024)

    float* out = (float*)d_out;
    float* out_logits = out;                                   // (B, L, VOCAB)
    float* out_hs     = out_logits + (size_t)NTOK * VOCAB;     // (NL, B, D_INNER, D_STATE)
    float* out_cache  = out_hs + (size_t)N_LAYERS * BATCH * D_INNER * D_STATE;

    // bump allocator over d_ws
    char* ws = (char*)d_ws;
    size_t off = 0;
    auto alloc = [&](size_t bytes) -> void* {
        off = (off + 255) & ~(size_t)255;
        void* p = ws + off;
        off += bytes;
        return p;
    };

    float*  x      = (float*) alloc((size_t)NTOK * D_MODEL * 4);
    float*  xn     = (float*) alloc((size_t)NTOK * D_MODEL * 4);
    __bf16* xn_bf  = (__bf16*)alloc((size_t)NTOK * D_MODEL * 2);
    float*  xz     = (float*) alloc((size_t)NTOK * 2 * D_INNER * 4);
    float*  xs     = (float*) alloc((size_t)NTOK * D_INNER * 4);
    __bf16* xs_bf  = (__bf16*)alloc((size_t)NTOK * D_INNER * 2);
    float*  dbl    = (float*) alloc((size_t)NTOK * (DT_RANK + 2 * D_STATE) * 4);
    __bf16* dbl_bf = (__bf16*)alloc((size_t)NTOK * (DT_RANK + 2 * D_STATE) * 2);
    float*  delta  = (float*) alloc((size_t)NTOK * D_INNER * 4);
    float*  yb     = (float*) alloc((size_t)NTOK * D_INNER * 4);
    __bf16* y_bf   = (__bf16*)alloc((size_t)NTOK * D_INNER * 2);
    __bf16* w_in   = (__bf16*)alloc((size_t)2 * D_INNER * D_MODEL * 2);
    __bf16* w_x    = (__bf16*)alloc((size_t)(DT_RANK + 2 * D_STATE) * D_INNER * 2);
    __bf16* w_dt   = (__bf16*)alloc((size_t)D_INNER * DT_RANK * 2);
    __bf16* w_out  = (__bf16*)alloc((size_t)D_MODEL * D_INNER * 2);
    __bf16* emb_bf = (__bf16*)alloc((size_t)VOCAB * D_MODEL * 2);

    auto cast = [&](const float* s, __bf16* d, long n) {
        int blocks = (int)((n + 255) / 256);
        if (blocks > 16384) blocks = 16384;
        k_cast_bf16<<<blocks, 256, 0, stream>>>(s, d, n);
    };

    // embedding table in bf16 (used by final logits GEMM)
    cast(embed, emb_bf, (long)VOCAB * D_MODEL);

    // x = embed[tokens]
    k_embed<<<NTOK, 256, 0, stream>>>(tokens, embed, x);

    for (int l = 0; l < N_LAYERS; ++l) {
        // rmsnorm + cast
        k_rmsnorm<<<NTOK, 256, 0, stream>>>(x, rms_w + (size_t)l * D_MODEL, xn, D_MODEL);
        cast(xn, xn_bf, (long)NTOK * D_MODEL);

        // in_proj: xz = xn * in_w^T   (M=2048, N=4096, K=1024)
        cast(in_proj_w + (size_t)l * 2 * D_INNER * D_MODEL, w_in, (long)2 * D_INNER * D_MODEL);
        k_gemm_bf16<<<gemm_grid(NTOK, 2 * D_INNER), 256, 0, stream>>>(
            xn_bf, w_in, nullptr, xz,
            NTOK, 2 * D_INNER, D_MODEL, D_MODEL, D_MODEL, 2 * D_INNER);

        // conv cache output + depthwise conv + silu
        k_conv_cache<<<(BATCH * D_INNER * (D_CONV - 1) + 255) / 256, 256, 0, stream>>>(
            xz, out_cache + (size_t)l * BATCH * D_INNER * (D_CONV - 1));
        k_conv_silu<<<(int)(((long)NTOK * D_INNER + 255) / 256), 256, 0, stream>>>(
            xz, conv_w + (size_t)l * D_INNER * D_CONV, conv_b + (size_t)l * D_INNER, xs);
        cast(xs, xs_bf, (long)NTOK * D_INNER);

        // x_proj: dbl = xs * xp_w^T   (M=2048, N=96, K=2048)
        cast(x_proj_w + (size_t)l * (DT_RANK + 2 * D_STATE) * D_INNER, w_x,
             (long)(DT_RANK + 2 * D_STATE) * D_INNER);
        k_gemm_bf16<<<gemm_grid(NTOK, DT_RANK + 2 * D_STATE), 256, 0, stream>>>(
            xs_bf, w_x, nullptr, dbl,
            NTOK, DT_RANK + 2 * D_STATE, D_INNER, D_INNER, D_INNER, DT_RANK + 2 * D_STATE);

        // dt_proj: draw = dbl[:, :64] * dt_w^T  (M=2048, N=2048, K=64, lda=96)
        cast(dbl, dbl_bf, (long)NTOK * (DT_RANK + 2 * D_STATE));
        cast(dt_proj_w + (size_t)l * D_INNER * DT_RANK, w_dt, (long)D_INNER * DT_RANK);
        k_gemm_bf16<<<gemm_grid(NTOK, D_INNER), 256, 0, stream>>>(
            dbl_bf, w_dt, nullptr, delta,
            NTOK, D_INNER, DT_RANK, DT_RANK + 2 * D_STATE, DT_RANK, D_INNER);

        // delta = softplus(draw + dt_b)
        k_softplus_bias<<<(int)(((long)NTOK * D_INNER + 255) / 256), 256, 0, stream>>>(
            delta, dt_proj_b + (size_t)l * D_INNER);

        // fused selective scan (writes hL into d_out)
        k_scan<<<BATCH * D_INNER / 128, 128, 0, stream>>>(
            delta, xs, dbl, xz,
            A_log + (size_t)l * D_INNER * D_STATE, D_param + (size_t)l * D_INNER,
            yb, out_hs + (size_t)l * BATCH * D_INNER * D_STATE);

        // out_proj + residual: x = y * out_w^T + x   (M=2048, N=1024, K=2048)
        cast(yb, y_bf, (long)NTOK * D_INNER);
        cast(out_proj_w + (size_t)l * D_MODEL * D_INNER, w_out, (long)D_MODEL * D_INNER);
        k_gemm_bf16<<<gemm_grid(NTOK, D_MODEL), 256, 0, stream>>>(
            y_bf, w_out, x, x,
            NTOK, D_MODEL, D_INNER, D_INNER, D_INNER, D_MODEL);
    }

    // final norm + logits
    k_rmsnorm<<<NTOK, 256, 0, stream>>>(x, norm_f_w, xn, D_MODEL);
    cast(xn, xn_bf, (long)NTOK * D_MODEL);
    k_gemm_bf16<<<gemm_grid(NTOK, VOCAB), 256, 0, stream>>>(
        xn_bf, emb_bf, nullptr, out_logits,
        NTOK, VOCAB, D_MODEL, D_MODEL, D_MODEL, VOCAB);
}
